// S5Layer_41120016892222
// MI455X (gfx1250) — matrix operand order repaired and checked by reading, not compiled, not run on hardware
//
#include <hip/hip_runtime.h>

typedef __bf16 bf16_t;
typedef __attribute__((ext_vector_type(16))) __bf16 v16bf;
typedef __attribute__((ext_vector_type(8)))  __bf16 v8bf;
typedef __attribute__((ext_vector_type(4)))  __bf16 v4bf;
typedef __attribute__((ext_vector_type(8)))  float  v8f;
typedef __attribute__((ext_vector_type(4)))  float  v4f;

#define L_SEQ  16384
#define H_DIM  1024
#define P_ST   512
#define N2P    1024      // 2*P (re|im stacked)
#define KDIM   1024
#define NKSTEP 32        // KDIM / 32
#define NCHUNK 128
#define TCHUNK 128
#define LDSP   40        // LDS row pitch in bf16 elems (32 + 8 pad, 80B rows keep 16B align)

// ---- gfx1250 async global->LDS path (guarded probe; falls back to sync) ----
#if defined(__has_builtin)
#if __has_builtin(__builtin_amdgcn_global_load_async_to_lds_b128)
#define ASYNC_LDS 1
#endif
#endif
#ifndef ASYNC_LDS
#define ASYNC_LDS 0
#endif

#if ASYNC_LDS
// prototype (from hipcc diagnostic): void(int4 as1*, int4 as3*, Ii offset, Ii cpol)
typedef int v4i_t __attribute__((vector_size(4 * sizeof(int))));
#define GCAST(p) ((__attribute__((address_space(1))) v4i_t*)(p))
#define LCAST(p) ((__attribute__((address_space(3))) v4i_t*)(p))
#if __has_builtin(__builtin_amdgcn_s_wait_asynccnt)
#define WAIT_ASYNC(n) __builtin_amdgcn_s_wait_asynccnt(n)
#else
#define WAIT_ASYNC(n) asm volatile("s_wait_asynccnt %0" :: "n"(n) : "memory")
#endif
#endif

// ---------------------------------------------------------------------------
// Prep: Lambda_bar; W1 = [Re(Bbar); Im(Bbar)] as (2P x H) bf16 (B-operand,
// N-major rows); W2 = [2*Cre | -2*Cim] stored N-major (h rows, 2P cols in K).
// ---------------------------------------------------------------------------
__global__ void prep_kernel(const float* __restrict__ Lre, const float* __restrict__ Lim,
                            const float* __restrict__ B,   const float* __restrict__ C,
                            const float* __restrict__ log_step,
                            bf16_t* __restrict__ W1, bf16_t* __restrict__ W2,
                            float* __restrict__ lam) {
  int id = blockIdx.x * blockDim.x + threadIdx.x;   // [0, P*H)
  // ---- W1 element (p, h) ----
  int p = id >> 10, h = id & (H_DIM - 1);
  float lre = Lre[p], lim = Lim[p];
  float dt  = __expf(log_step[p]);
  float ear = __expf(lre * dt);
  float s, c; __sincosf(lim * dt, &s, &c);
  float lamr = ear * c, lami = ear * s;             // Lambda_bar
  float nr = lamr - 1.0f, ni = lami;                // (Lambda_bar-1)/Lambda
  float inv = 1.0f / (lre * lre + lim * lim);
  float cr = (nr * lre + ni * lim) * inv;
  float ci = (ni * lre - nr * lim) * inv;
  float br = B[(size_t)(p * H_DIM + h) * 2 + 0];
  float bi = B[(size_t)(p * H_DIM + h) * 2 + 1];
  W1[(size_t)p * H_DIM + h]          = (bf16_t)(cr * br - ci * bi);   // Re(Bbar)
  W1[(size_t)(P_ST + p) * H_DIM + h] = (bf16_t)(cr * bi + ci * br);   // Im(Bbar)
  if (h == 0) { lam[2 * p] = lamr; lam[2 * p + 1] = lami; }
  // ---- W2 element (h2, p2) ----
  int h2 = id >> 9, p2 = id & (P_ST - 1);
  float c_re = C[(size_t)(h2 * P_ST + p2) * 2 + 0];
  float c_im = C[(size_t)(h2 * P_ST + p2) * 2 + 1];
  W2[(size_t)h2 * N2P + p2]        = (bf16_t)( 2.0f * c_re);
  W2[(size_t)h2 * N2P + P_ST + p2] = (bf16_t)(-2.0f * c_im);
}

// ---------------------------------------------------------------------------
// WMMA GEMM: Out[M x 1024] = A[M x 1024] * B[1024 x 1024]
//   A row-major (f32 converted inline, or bf16); Bcm = B^T row-major.
// Block tile 128x128, 8 waves (2x4), each wave 64x32 = 4x2 WMMA tiles.
// Double-buffered LDS; bf16 tiles staged via async global->LDS when present.
// ---------------------------------------------------------------------------
template<bool A_F32, bool EPI>
__global__ void __launch_bounds__(256) gemm_kernel(const void* __restrict__ Aptr,
                                                   const bf16_t* __restrict__ Bcm,
                                                   float* __restrict__ Out,
                                                   const float* __restrict__ U,
                                                   const float* __restrict__ Dv) {
  __shared__ __align__(16) bf16_t sA[2][128 * LDSP];
  __shared__ __align__(16) bf16_t sB[2][128 * LDSP];
  const int tid  = threadIdx.x;
  const int lane = tid & 31, wid = tid >> 5;
  const int wm = wid >> 2, wn = wid & 3;            // 2 x 4 wave grid
  const int row0 = blockIdx.y * 128;
  const int col0 = blockIdx.x * 128;
  const int nlan = lane & 15, sel = lane >> 4;

  v8f acc[4][2];
#pragma unroll
  for (int mt = 0; mt < 4; ++mt)
#pragma unroll
    for (int nt = 0; nt < 2; ++nt)
#pragma unroll
      for (int e = 0; e < 8; ++e) acc[mt][nt][e] = 0.0f;

  // per-thread staging coordinates
  const int r2 = tid >> 2, c2 = (tid & 3) * 8;      // bf16 tile chunks (i=0,1 adds 64 rows)
  v4f  regA4[4];                                    // A f32 staging regs
#if !ASYNC_LDS
  v8bf regA8[2];                                    // A bf16 staging regs (sync path)
  v8bf regB8[2];                                    // B staging regs (sync path)
#endif
#if ASYNC_LDS
  constexpr unsigned short NASY = A_F32 ? 2 : 4;    // asyncs issued per thread per tile
#endif

  // ---- prologue: fetch tile 0 ----
  if (A_F32) {
    const float* A = (const float*)Aptr;
#pragma unroll
    for (int i = 0; i < 4; ++i) {
      int idx = tid + 256 * i;
      regA4[i] = *(const v4f*)(A + (size_t)(row0 + (idx >> 3)) * KDIM + (idx & 7) * 4);
    }
  }
#if ASYNC_LDS
  if (!A_F32) {
    const bf16_t* A = (const bf16_t*)Aptr;
#pragma unroll
    for (int i = 0; i < 2; ++i)
      __builtin_amdgcn_global_load_async_to_lds_b128(
          GCAST(A + (size_t)(row0 + r2 + 64 * i) * KDIM + c2),
          LCAST(&sA[0][(r2 + 64 * i) * LDSP + c2]), 0, 0);
  }
#pragma unroll
  for (int i = 0; i < 2; ++i)
    __builtin_amdgcn_global_load_async_to_lds_b128(
        GCAST(Bcm + (size_t)(col0 + r2 + 64 * i) * KDIM + c2),
        LCAST(&sB[0][(r2 + 64 * i) * LDSP + c2]), 0, 0);
#else
  if (!A_F32) {
    const bf16_t* A = (const bf16_t*)Aptr;
#pragma unroll
    for (int i = 0; i < 2; ++i)
      regA8[i] = *(const v8bf*)(A + (size_t)(row0 + r2 + 64 * i) * KDIM + c2);
  }
#pragma unroll
  for (int i = 0; i < 2; ++i)
    regB8[i] = *(const v8bf*)(Bcm + (size_t)(col0 + r2 + 64 * i) * KDIM + c2);
#endif

  for (int kt = 0; kt < NKSTEP; ++kt) {
    const int cur = kt & 1;
    const int k0  = kt * 32;
    // ---- stage current-tile regs into LDS (non-async parts) ----
    if (A_F32) {
#pragma unroll
      for (int i = 0; i < 4; ++i) {
        int idx = tid + 256 * i;
        v4bf b; b[0] = (bf16_t)regA4[i].x; b[1] = (bf16_t)regA4[i].y;
        b[2] = (bf16_t)regA4[i].z; b[3] = (bf16_t)regA4[i].w;
        *(v4bf*)&sA[cur][(idx >> 3) * LDSP + (idx & 7) * 4] = b;
      }
    }
#if !ASYNC_LDS
    if (!A_F32) {
#pragma unroll
      for (int i = 0; i < 2; ++i)
        *(v8bf*)&sA[cur][(r2 + 64 * i) * LDSP + c2] = regA8[i];
    }
#pragma unroll
    for (int i = 0; i < 2; ++i)
      *(v8bf*)&sB[cur][(r2 + 64 * i) * LDSP + c2] = regB8[i];
#endif
    // ---- issue next-tile fetches ----
    if (kt + 1 < NKSTEP) {
      const int kn = k0 + 32, nxt = cur ^ 1;
      if (A_F32) {
        const float* A = (const float*)Aptr;
#pragma unroll
        for (int i = 0; i < 4; ++i) {
          int idx = tid + 256 * i;
          regA4[i] = *(const v4f*)(A + (size_t)(row0 + (idx >> 3)) * KDIM + kn + (idx & 7) * 4);
        }
      }
#if ASYNC_LDS
      if (!A_F32) {
        const bf16_t* A = (const bf16_t*)Aptr;
#pragma unroll
        for (int i = 0; i < 2; ++i)
          __builtin_amdgcn_global_load_async_to_lds_b128(
              GCAST(A + (size_t)(row0 + r2 + 64 * i) * KDIM + kn + c2),
              LCAST(&sA[nxt][(r2 + 64 * i) * LDSP + c2]), 0, 0);
      }
#pragma unroll
      for (int i = 0; i < 2; ++i)
        __builtin_amdgcn_global_load_async_to_lds_b128(
            GCAST(Bcm + (size_t)(col0 + r2 + 64 * i) * KDIM + kn + c2),
            LCAST(&sB[nxt][(r2 + 64 * i) * LDSP + c2]), 0, 0);
#else
      if (!A_F32) {
        const bf16_t* A = (const bf16_t*)Aptr;
#pragma unroll
        for (int i = 0; i < 2; ++i)
          regA8[i] = *(const v8bf*)(A + (size_t)(row0 + r2 + 64 * i) * KDIM + kn + c2);
      }
#pragma unroll
      for (int i = 0; i < 2; ++i)
        regB8[i] = *(const v8bf*)(Bcm + (size_t)(col0 + r2 + 64 * i) * KDIM + kn + c2);
#endif
    }
#if ASYNC_LDS
    // in-order async completion: <=NASY outstanding => current tile resident
    if (kt + 1 < NKSTEP) WAIT_ASYNC(NASY); else WAIT_ASYNC(0);
#endif
    __syncthreads();

    // ---- fragments per ISA 7.12.2 ----
    v16bf afrag[4], bfrag[2];
#pragma unroll
    for (int mt = 0; mt < 4; ++mt) {
      // A 16x32: lanes 0-15 rows m, K 0-7 / 16-23; lanes 16-31 K 8-15 / 24-31
      const bf16_t* base = &sA[cur][(wm * 64 + mt * 16 + nlan) * LDSP + sel * 8];
      v8bf lo = *(const v8bf*)base;
      v8bf hi = *(const v8bf*)(base + 16);
      afrag[mt] = __builtin_shufflevector(lo, hi, 0,1,2,3,4,5,6,7,8,9,10,11,12,13,14,15);
    }
#pragma unroll
    for (int nt = 0; nt < 2; ++nt) {
      // B 32x16: lanes 0-15 column n, K 0-7 / 16-23; lanes 16-31 K 8-15 / 24-31
      const bf16_t* base = &sB[cur][(wn * 32 + nt * 16 + nlan) * LDSP + sel * 8];
      v8bf lo = *(const v8bf*)base;
      v8bf hi = *(const v8bf*)(base + 16);
      bfrag[nt] = __builtin_shufflevector(lo, hi, 0,1,2,3,4,5,6,7,8,9,10,11,12,13,14,15);
    }
#pragma unroll
    for (int mt = 0; mt < 4; ++mt)
#pragma unroll
      for (int nt = 0; nt < 2; ++nt)
        acc[mt][nt] = __builtin_amdgcn_wmma_f32_16x16x32_bf16(
            false, afrag[mt], false, bfrag[nt], (short)0, acc[mt][nt], false, false);
    __syncthreads();
  }

  // ---- store C (VGPR r: lanes 0-15 -> M=r, lanes 16-31 -> M=8+r; N=lane&15)
#pragma unroll
  for (int mt = 0; mt < 4; ++mt)
#pragma unroll
    for (int nt = 0; nt < 2; ++nt) {
      int col = col0 + wn * 32 + nt * 16 + nlan;
      float dv = EPI ? Dv[col] : 0.0f;
#pragma unroll
      for (int r = 0; r < 8; ++r) {
        int row = row0 + wm * 64 + mt * 16 + sel * 8 + r;
        float v = acc[mt][nt][r];
        if (EPI) v += dv * U[(size_t)row * H_DIM + col];
        Out[(size_t)row * N2P + col] = v;
      }
    }
}

// ---------------------------------------------------------------------------
// Chunked complex scan: x_l = a*x_{l-1} + Bu_l   (a = Lambda_bar[p])
// ---------------------------------------------------------------------------
__global__ void scan_local(float* __restrict__ Bu, const float* __restrict__ lam,
                           float* __restrict__ S) {
  int c = blockIdx.x >> 1;
  int p = ((blockIdx.x & 1) << 8) + threadIdx.x;
  float ar = lam[2 * p], ai = lam[2 * p + 1];
  float xr = 0.0f, xi = 0.0f;
  size_t base = (size_t)c * TCHUNK * N2P + p;
  for (int t = 0; t < TCHUNK; ++t) {
    float br = Bu[base], bi = Bu[base + P_ST];
    float nxr = ar * xr - ai * xi + br;
    float nxi = ar * xi + ai * xr + bi;
    xr = nxr; xi = nxi;
    Bu[base] = xr; Bu[base + P_ST] = xi;
    base += N2P;
  }
  S[(size_t)(c * P_ST + p) * 2 + 0] = xr;
  S[(size_t)(c * P_ST + p) * 2 + 1] = xi;
}

__global__ void scan_chunks(const float* __restrict__ S, const float* __restrict__ lam,
                            float* __restrict__ Sprev) {
  int p = threadIdx.x;                               // 512 threads
  float ar = lam[2 * p], ai = lam[2 * p + 1];
  float tr = ar, ti = ai;                            // a^128 via 7 squarings
  for (int i = 0; i < 7; ++i) { float q = tr * tr - ti * ti; ti = 2.0f * tr * ti; tr = q; }
  float rr = 0.0f, ri = 0.0f;
  for (int c = 0; c < NCHUNK; ++c) {
    Sprev[(size_t)(c * P_ST + p) * 2 + 0] = rr;
    Sprev[(size_t)(c * P_ST + p) * 2 + 1] = ri;
    float sr = S[(size_t)(c * P_ST + p) * 2 + 0];
    float si = S[(size_t)(c * P_ST + p) * 2 + 1];
    float q0 = tr * rr - ti * ri + sr;
    float q1 = tr * ri + ti * rr + si;
    rr = q0; ri = q1;
  }
}

__global__ void scan_apply(const float* __restrict__ Bu, const float* __restrict__ lam,
                           const float* __restrict__ Sprev, bf16_t* __restrict__ XS) {
  int c = blockIdx.x >> 1;
  int p = ((blockIdx.x & 1) << 8) + threadIdx.x;
  float ar = lam[2 * p], ai = lam[2 * p + 1];
  float sr = Sprev[(size_t)(c * P_ST + p) * 2 + 0];
  float si = Sprev[(size_t)(c * P_ST + p) * 2 + 1];
  float wr = ar, wi = ai;
  size_t base = (size_t)c * TCHUNK * N2P + p;
  for (int t = 0; t < TCHUNK; ++t) {
    float xr = Bu[base]        + (wr * sr - wi * si);
    float xi = Bu[base + P_ST] + (wr * si + wi * sr);
    XS[base]        = (bf16_t)xr;
    XS[base + P_ST] = (bf16_t)xi;
    float q0 = wr * ar - wi * ai, q1 = wr * ai + wi * ar;
    wr = q0; wi = q1;
    base += N2P;
  }
}

// ---------------------------------------------------------------------------
extern "C" void kernel_launch(void* const* d_in, const int* in_sizes, int n_in,
                              void* d_out, int out_size, void* d_ws, size_t ws_size,
                              hipStream_t stream) {
  (void)in_sizes; (void)n_in; (void)out_size; (void)ws_size;
  const float* u   = (const float*)d_in[0];
  const float* Lre = (const float*)d_in[1];
  const float* Lim = (const float*)d_in[2];
  const float* B   = (const float*)d_in[3];
  const float* C   = (const float*)d_in[4];
  const float* D   = (const float*)d_in[5];
  const float* ls  = (const float*)d_in[6];
  float* out = (float*)d_out;

  char* ws = (char*)d_ws;
  bf16_t* W1  = (bf16_t*)(ws);                                   //  2 MB
  bf16_t* W2  = (bf16_t*)(ws + ((size_t)2  << 20));              //  2 MB
  float*  Bu  = (float*) (ws + ((size_t)4  << 20));              // 64 MB
  bf16_t* XS  = (bf16_t*)(ws + ((size_t)68 << 20));              // 32 MB
  float*  lam = (float*) (ws + ((size_t)100 << 20));             //  4 KB
  float*  S   = (float*) (ws + ((size_t)100 << 20) + (1u << 16));// 512 KB
  float*  Sp  = (float*) (ws + ((size_t)100 << 20) + (1u << 16) + (1u << 19));

  prep_kernel<<<2048, 256, 0, stream>>>(Lre, Lim, B, C, ls, W1, W2, lam);
  gemm_kernel<true,  false><<<dim3(N2P / 128, L_SEQ / 128), 256, 0, stream>>>(u,  W1, Bu,  nullptr, nullptr);
  scan_local <<<2 * NCHUNK, 256, 0, stream>>>(Bu, lam, S);
  scan_chunks<<<1, P_ST, 0, stream>>>(S, lam, Sp);
  scan_apply <<<2 * NCHUNK, 256, 0, stream>>>(Bu, lam, Sp, XS);
  gemm_kernel<false, true ><<<dim3(H_DIM / 128, L_SEQ / 128), 256, 0, stream>>>(XS, W2, out, u, D);
}
